// VerGDTransform_39195871543394
// MI455X (gfx1250) — matrix operand order repaired and checked by reading, not compiled, not run on hardware
//
#include <hip/hip_runtime.h>
#include <hip/hip_bf16.h>

// ---------------------------------------------------------------------------
// VerGDTransform on MI455X (gfx1250): gather -> MLP(129->256->128) ->
// segment-sum(8 consecutive rows) -> MLP(128->256->128).
// Compute-bound (~118 GFLOP vs ~140MB traffic) => bf16 WMMA 16x16x32, f32 acc.
// ---------------------------------------------------------------------------

typedef __attribute__((ext_vector_type(16))) __bf16        v16bf;
typedef __attribute__((ext_vector_type(8)))  float         v8f;
typedef __attribute__((ext_vector_type(8)))  unsigned int  v8u;

#define N_GROUPS   100000
#define PER_GROUP  8
#define E_TOTAL    (N_GROUPS * PER_GROUP)
#define EMB        128
#define HID        256
#define MT_ROWS    64          // rows per workgroup tile (8 groups)

// round-to-nearest-even f32 -> bf16 (as ushort)
__device__ __forceinline__ unsigned short f2bf(float f) {
    unsigned int u = __builtin_bit_cast(unsigned int, f);
    u += 0x7FFFu + ((u >> 16) & 1u);
    return (unsigned short)(u >> 16);
}

// A-fragment from LDS (row-major bf16): e0..7 at p, e8..15 at p+16 elements.
__device__ __forceinline__ v16bf load_frag_s(const unsigned short* p) {
    uint4 lo = *(const uint4*)p;
    uint4 hi = *(const uint4*)(p + 16);
    v8u u = {lo.x, lo.y, lo.z, lo.w, hi.x, hi.y, hi.z, hi.w};
    return __builtin_bit_cast(v16bf, u);
}

// B-fragment from packed global weights: 32 contiguous bytes per lane.
__device__ __forceinline__ v16bf load_frag_g(const unsigned short* p) {
    uint4 lo = ((const uint4*)p)[0];
    uint4 hi = ((const uint4*)p)[1];
    v8u u = {lo.x, lo.y, lo.z, lo.w, hi.x, hi.y, hi.z, hi.w};
    return __builtin_bit_cast(v16bf, u);
}

// ---------------------------------------------------------------------------
// Pack f32 weights into bf16 B-fragment order.
// W1-type [128,HID]:  packedIdx = ((kt*16 + nt)*32 + lane)*16 + e
//   K = kt*32 + (lane>>4)*8 + (e&7) + (e>>3)*16 ; N = nt*16 + (lane&15)
// W2-type [HID,128]:  packedIdx = ((kt*8 + nt)*32 + lane)*16 + e  (8 kt, 8 nt)
// ---------------------------------------------------------------------------
__global__ __launch_bounds__(256) void k_pack(
    const float* __restrict__ W1c, const float* __restrict__ W2c,
    const float* __restrict__ W1p, const float* __restrict__ W2p,
    unsigned short* __restrict__ W1cP, unsigned short* __restrict__ W2cP,
    unsigned short* __restrict__ W1pP, unsigned short* __restrict__ W2pP)
{
    int t = blockIdx.x * 256 + threadIdx.x;      // 0 .. 131071
    int which = t >> 15;                          // 0:W1c 1:W2c 2:W1p 3:W2p
    int idx = t & 32767;
    int lane = (idx >> 4) & 31;
    int e    = idx & 15;
    int kInner = ((lane >> 4) << 3) + (e & 7) + ((e >> 3) << 4);  // (lane/16)*8 + (e&7) + (e/8)*16
    int nLane  = lane & 15;

    if (which == 0 || which == 2) {               // K=128, N=256
        int kt = idx >> 13;                       // 4 k-tiles
        int nt = (idx >> 9) & 15;                 // 16 n-tiles
        int K = kt * 32 + kInner;
        int N = nt * 16 + nLane;
        const float* src = (which == 0) ? W1c : W1p;
        unsigned short* dst = (which == 0) ? W1cP : W1pP;
        dst[idx] = f2bf(src[K * HID + N]);
    } else {                                      // K=256, N=128
        int kt = idx >> 12;                       // 8 k-tiles
        int nt = (idx >> 9) & 7;                  // 8 n-tiles
        int K = kt * 32 + kInner;
        int N = nt * 16 + nLane;
        const float* src = (which == 1) ? W2c : W2p;
        unsigned short* dst = (which == 1) ? W2cP : W2pP;
        dst[idx] = f2bf(src[K * EMB + N]);
    }
}

// ---------------------------------------------------------------------------
// Stage 1: gather + MLP_combine + segment-sum of 8 consecutive rows.
// One block = 64 E-rows = 8 groups. 8 waves.
// ---------------------------------------------------------------------------
__global__ __launch_bounds__(256) void k_stage1(
    const float* __restrict__ repr, const int* __restrict__ gd,
    const float* __restrict__ gd_deg,
    const float* __restrict__ W1c,   // f32, row 128 is the deg row
    const float* __restrict__ b1c, const float* __restrict__ b2c,
    const unsigned short* __restrict__ W1P, const unsigned short* __restrict__ W2P,
    float* __restrict__ seg)
{
    __shared__ __align__(16) unsigned char smemRaw[51200];
    __shared__ float degs[MT_ROWS];
    unsigned short (*Xs)[136] = (unsigned short(*)[136])smemRaw;              // 17408 B
    unsigned short (*Hs)[264] = (unsigned short(*)[264])(smemRaw + 17408);    // 33792 B
    float          (*Ys)[132] = (float(*)[132])smemRaw;                       // overlay

    const int t   = threadIdx.x;
    const int blk = blockIdx.x;
    const int m0  = blk * MT_ROWS;

    // ---- gather + f32->bf16 convert into LDS: 4 threads per row -----------
    {
        int r  = t >> 2;
        int c0 = (t & 3) * 32;
        int m  = m0 + r;
        int node = gd[m];
        const float* src = repr + (long)node * EMB + c0;
        unsigned int* dst = (unsigned int*)&Xs[r][c0];
        #pragma unroll
        for (int j = 0; j < 8; ++j) {
            float4 v = ((const float4*)src)[j];
            dst[2*j]   = (unsigned)f2bf(v.x) | ((unsigned)f2bf(v.y) << 16);
            dst[2*j+1] = (unsigned)f2bf(v.z) | ((unsigned)f2bf(v.w) << 16);
        }
        if ((t & 3) == 0) degs[r] = gd_deg[m];
    }
    __syncthreads();

    const int w     = t >> 5;
    const int lane  = t & 31;
    const int mrow  = lane & 15;
    const int khalf = (lane >> 4) * 8;

    // ---- GEMM1: X[64,128] x W1[128,256] -> H[64,256], wave w owns nt=2w,2w+1
    v8f acc[4][2] = {};
    #pragma unroll
    for (int kt = 0; kt < 4; ++kt) {
        const int kb = kt * 32 + khalf;
        v16bf a[4];
        #pragma unroll
        for (int mt = 0; mt < 4; ++mt)
            a[mt] = load_frag_s(&Xs[mt * 16 + mrow][kb]);
        #pragma unroll
        for (int ntl = 0; ntl < 2; ++ntl) {
            const int nt = w * 2 + ntl;
            v16bf b = load_frag_g(&W1P[((kt * 16 + nt) * 32 + lane) << 4]);
            #pragma unroll
            for (int mt = 0; mt < 4; ++mt)
                acc[mt][ntl] = __builtin_amdgcn_wmma_f32_16x16x32_bf16(
                    false, a[mt], false, b, (short)0, acc[mt][ntl], false, false);
        }
    }

    // ---- epilogue 1: + b1 + deg * W1c[128,:], ReLU, bf16 -> Hs ------------
    {
        const float* w1deg = W1c + 128 * HID;
        #pragma unroll
        for (int ntl = 0; ntl < 2; ++ntl) {
            const int col = (w * 2 + ntl) * 16 + mrow;
            const float bb = b1c[col];
            const float wd = w1deg[col];
            #pragma unroll
            for (int mt = 0; mt < 4; ++mt) {
                #pragma unroll
                for (int rr = 0; rr < 8; ++rr) {
                    const int row = mt * 16 + khalf + rr;
                    float h = acc[mt][ntl][rr] + bb + degs[row] * wd;
                    Hs[row][col] = f2bf(fmaxf(h, 0.0f));
                }
            }
        }
    }
    __syncthreads();

    // ---- GEMM2: H[64,256] x W2[256,128] -> Y[64,128], wave w owns nt=w ----
    v8f acc2[4] = {};
    #pragma unroll
    for (int kt = 0; kt < 8; ++kt) {
        const int kb = kt * 32 + khalf;
        v16bf a[4];
        #pragma unroll
        for (int mt = 0; mt < 4; ++mt)
            a[mt] = load_frag_s(&Hs[mt * 16 + mrow][kb]);
        v16bf b = load_frag_g(&W2P[((kt * 8 + w) * 32 + lane) << 4]);
        #pragma unroll
        for (int mt = 0; mt < 4; ++mt)
            acc2[mt] = __builtin_amdgcn_wmma_f32_16x16x32_bf16(
                false, a[mt], false, b, (short)0, acc2[mt], false, false);
    }
    __syncthreads();   // everyone done reading Xs/Hs -> Ys may overlay

    // ---- epilogue 2: + b2, spill Y to LDS for segment reduction -----------
    {
        const int col = w * 16 + mrow;
        const float bb = b2c[col];
        #pragma unroll
        for (int mt = 0; mt < 4; ++mt)
            #pragma unroll
            for (int rr = 0; rr < 8; ++rr)
                Ys[mt * 16 + khalf + rr][col] = acc2[mt][rr] + bb;
    }
    __syncthreads();

    // ---- segment sum: group g <- rows 8g..8g+7; 8 groups x 128 cols -------
    #pragma unroll
    for (int i = 0; i < 4; ++i) {
        int idx = t + i * 256;          // 0..1023
        int g = idx >> 7;
        int c = idx & 127;
        float s = 0.0f;
        #pragma unroll
        for (int rr = 0; rr < 8; ++rr) s += Ys[g * 8 + rr][c];
        seg[(long)(blk * 8 + g) * EMB + c] = s;
    }
}

// ---------------------------------------------------------------------------
// Stage 2: out = relu(seg @ W1p + b1p) @ W2p + b2p. 64 groups per block.
// ---------------------------------------------------------------------------
__global__ __launch_bounds__(256) void k_stage2(
    const float* __restrict__ seg,
    const float* __restrict__ b1p, const float* __restrict__ b2p,
    const unsigned short* __restrict__ W1P, const unsigned short* __restrict__ W2P,
    float* __restrict__ out, int G)
{
    __shared__ __align__(16) unsigned char smemRaw[51200];
    unsigned short (*Xs)[136] = (unsigned short(*)[136])smemRaw;
    unsigned short (*Hs)[264] = (unsigned short(*)[264])(smemRaw + 17408);

    const int t   = threadIdx.x;
    const int blk = blockIdx.x;
    const int g0  = blk * MT_ROWS;

    // ---- load seg tile -> bf16 LDS (zero-fill past G) ---------------------
    {
        int r  = t >> 2;
        int c0 = (t & 3) * 32;
        int g  = g0 + r;
        unsigned int* dst = (unsigned int*)&Xs[r][c0];
        if (g < G) {
            const float* src = seg + (long)g * EMB + c0;
            #pragma unroll
            for (int j = 0; j < 8; ++j) {
                float4 v = ((const float4*)src)[j];
                dst[2*j]   = (unsigned)f2bf(v.x) | ((unsigned)f2bf(v.y) << 16);
                dst[2*j+1] = (unsigned)f2bf(v.z) | ((unsigned)f2bf(v.w) << 16);
            }
        } else {
            #pragma unroll
            for (int j = 0; j < 16; ++j) dst[j] = 0u;
        }
    }
    __syncthreads();

    const int w     = t >> 5;
    const int lane  = t & 31;
    const int mrow  = lane & 15;
    const int khalf = (lane >> 4) * 8;

    // ---- GEMM1 ------------------------------------------------------------
    v8f acc[4][2] = {};
    #pragma unroll
    for (int kt = 0; kt < 4; ++kt) {
        const int kb = kt * 32 + khalf;
        v16bf a[4];
        #pragma unroll
        for (int mt = 0; mt < 4; ++mt)
            a[mt] = load_frag_s(&Xs[mt * 16 + mrow][kb]);
        #pragma unroll
        for (int ntl = 0; ntl < 2; ++ntl) {
            const int nt = w * 2 + ntl;
            v16bf b = load_frag_g(&W1P[((kt * 16 + nt) * 32 + lane) << 4]);
            #pragma unroll
            for (int mt = 0; mt < 4; ++mt)
                acc[mt][ntl] = __builtin_amdgcn_wmma_f32_16x16x32_bf16(
                    false, a[mt], false, b, (short)0, acc[mt][ntl], false, false);
        }
    }

    // ---- epilogue 1: + b1p, ReLU, bf16 -> Hs ------------------------------
    #pragma unroll
    for (int ntl = 0; ntl < 2; ++ntl) {
        const int col = (w * 2 + ntl) * 16 + mrow;
        const float bb = b1p[col];
        #pragma unroll
        for (int mt = 0; mt < 4; ++mt)
            #pragma unroll
            for (int rr = 0; rr < 8; ++rr) {
                const int row = mt * 16 + khalf + rr;
                Hs[row][col] = f2bf(fmaxf(acc[mt][ntl][rr] + bb, 0.0f));
            }
    }
    __syncthreads();

    // ---- GEMM2 ------------------------------------------------------------
    v8f acc2[4] = {};
    #pragma unroll
    for (int kt = 0; kt < 8; ++kt) {
        const int kb = kt * 32 + khalf;
        v16bf a[4];
        #pragma unroll
        for (int mt = 0; mt < 4; ++mt)
            a[mt] = load_frag_s(&Hs[mt * 16 + mrow][kb]);
        v16bf b = load_frag_g(&W2P[((kt * 8 + w) * 32 + lane) << 4]);
        #pragma unroll
        for (int mt = 0; mt < 4; ++mt)
            acc2[mt] = __builtin_amdgcn_wmma_f32_16x16x32_bf16(
                false, a[mt], false, b, (short)0, acc2[mt], false, false);
    }

    // ---- epilogue 2: + b2p, guarded store ---------------------------------
    {
        const int col = w * 16 + mrow;
        const float bb = b2p[col];
        #pragma unroll
        for (int mt = 0; mt < 4; ++mt)
            #pragma unroll
            for (int rr = 0; rr < 8; ++rr) {
                const int row = mt * 16 + khalf + rr;
                const int g = g0 + row;
                if (g < G) out[(long)g * EMB + col] = acc2[mt][rr] + bb;
            }
    }
}

// ---------------------------------------------------------------------------
extern "C" void kernel_launch(void* const* d_in, const int* in_sizes, int n_in,
                              void* d_out, int out_size, void* d_ws, size_t ws_size,
                              hipStream_t stream) {
    (void)in_sizes; (void)n_in; (void)out_size; (void)ws_size;

    // inputs in setup_inputs() order
    const float* repr   = (const float*)d_in[0];
    const int*   gd     = (const int*)  d_in[1];
    /* gd_len (always 8) = d_in[2] — groups are consecutive blocks of 8 */
    const float* gd_deg = (const float*)d_in[3];
    const float* W1c    = (const float*)d_in[4];   // [129,256]
    const float* b1c    = (const float*)d_in[5];
    const float* W2c    = (const float*)d_in[6];   // [256,128]
    const float* b2c    = (const float*)d_in[7];
    const float* W1p    = (const float*)d_in[8];   // [128,256]
    const float* b1p    = (const float*)d_in[9];
    const float* W2p    = (const float*)d_in[10];  // [256,128]
    const float* b2p    = (const float*)d_in[11];
    float* out = (float*)d_out;

    // workspace layout
    float* seg = (float*)d_ws;                                  // [G,128] f32
    unsigned short* W1cP = (unsigned short*)(seg + (size_t)N_GROUPS * EMB);
    unsigned short* W2cP = W1cP + 32768;
    unsigned short* W1pP = W2cP + 32768;
    unsigned short* W2pP = W1pP + 32768;

    // 1) pack weights f32 -> bf16 fragment order (131072 elements)
    k_pack<<<512, 256, 0, stream>>>(W1c, W2c, W1p, W2p, W1cP, W2cP, W1pP, W2pP);

    // 2) gather + MLP_combine + segment-sum  (800000 / 64 = 12500 blocks)
    k_stage1<<<E_TOTAL / MT_ROWS, 256, 0, stream>>>(
        repr, gd, gd_deg, W1c, b1c, b2c, W1cP, W2cP, seg);

    // 3) MLP_process on seg  (ceil(100000/64) = 1563 blocks)
    k_stage2<<<(N_GROUPS + MT_ROWS - 1) / MT_ROWS, 256, 0, stream>>>(
        seg, b1p, b2p, W1pP, W2pP, out, N_GROUPS);
}
